// RNN_15410342658561
// MI455X (gfx1250) — matrix operand order repaired and checked by reading, not compiled, not run on hardware
//
#include <hip/hip_runtime.h>

// Problem constants (match reference)
#define B_ 32
#define S_ 512
#define V_ 8000
#define I_ 256
#define H_ 512
#define LN_EPS 1e-5f

typedef __attribute__((ext_vector_type(16))) _Float16 v16h;
typedef __attribute__((ext_vector_type(8)))  float    v8f;
typedef __attribute__((ext_vector_type(4)))  unsigned int u32x4;
typedef __attribute__((ext_vector_type(8)))  int          i32x8;
typedef __attribute__((ext_vector_type(4)))  int          i32x4;

union Frag16 {
    v16h  v;
    uint4 q[2];
};

union Pack32 {
    _Float16 h[32];
    uint4    q[4];
};

// Load a 16-element f16 fragment as two 16B chunks (generic ptr: works for LDS & global)
static __device__ inline v16h load_frag(const _Float16* p0, const _Float16* p1) {
    Frag16 f;
    f.q[0] = *(const uint4*)p0;
    f.q[1] = *(const uint4*)p1;
    return f.v;
}

static __device__ inline v8f wmma_f16(v16h a, v16h b, v8f c) {
    // D = A(16x32 f16) * B(32x16 f16) + C(16x16 f32)
    return __builtin_amdgcn_wmma_f32_16x16x32_f16(
        /*neg_a=*/false, a, /*neg_b=*/false, b,
        /*c_mod=*/(short)0, c, /*reuse_a=*/false, /*reuse_b=*/false);
}

// ---------------------------------------------------------------------------
// Tensor Data Mover: 2D tile load (global f16 matrix -> packed LDS tile).
// D# built per CDNA5 ISA ch.8: group0 = {flags, lds_addr, global_addr, type},
// group1 = {mask/data_size, tensor dims, tile dims, stride0}. 2D => groups 2/3
// (and the trailing group) zero. This toolchain declares the 6-arg builtin:
// (uint32x4, int32x8, int32x4, int32x4, int32x8, i32 cpol).
// ---------------------------------------------------------------------------
static __device__ inline void tdm_load_2d(unsigned int lds_addr, const void* gptr,
                                          unsigned int tensor_d0, unsigned int tensor_d1,
                                          unsigned int tile_d0,  unsigned int tile_d1,
                                          unsigned int stride0_elems) {
    unsigned long long ga = (unsigned long long)(uintptr_t)gptr;
    u32x4 g0;
    g0[0] = 1u;                                               // count=1, user mode
    g0[1] = lds_addr;                                         // lds_addr [63:32]
    g0[2] = (unsigned int)ga;                                 // global_addr[31:0]
    g0[3] = (unsigned int)((ga >> 32) & 0x1FFFFFFu) | (2u << 30); // addr[56:32] | type=2
    i32x8 g1;
    g1[0] = (int)(1u << 16);                                  // data_size=1 (2 bytes)
    g1[1] = (int)(tensor_d0 << 16);                           // tensor_dim0[15:0] @ bit48
    g1[2] = (int)((tensor_d0 >> 16) | (tensor_d1 << 16));     // d0 hi | d1 lo
    g1[3] = (int)((tensor_d1 >> 16) | (tile_d0 << 16));       // d1 hi | tile_dim0
    g1[4] = (int)(tile_d1);                                   // tile_dim1 (tile_dim2=0)
    g1[5] = (int)stride0_elems;                               // tensor_dim0_stride[31:0]
    g1[6] = 0;                                                // stride0 hi | stride1 lo
    g1[7] = 0;
    i32x4 z4 = {0, 0, 0, 0};
    i32x8 z8 = {0, 0, 0, 0, 0, 0, 0, 0};
    __builtin_amdgcn_tensor_load_to_lds(g0, g1, z4, z4, z8, 0);
}

// ---------------------------------------------------------------------------
// f32 -> f16 conversion (weights)
// ---------------------------------------------------------------------------
__global__ void cvt_f32_f16(const float* __restrict__ src, _Float16* __restrict__ dst, int n) {
    int i = blockIdx.x * blockDim.x + threadIdx.x;
    if (i < n) dst[i] = (_Float16)src[i];
}

// ---------------------------------------------------------------------------
// Embedding gather: xs[s][b][i] = (f16) emb[x_idx[b][s]][i]
// ---------------------------------------------------------------------------
__global__ void gather_emb(const int* __restrict__ idx, const float* __restrict__ emb,
                           _Float16* __restrict__ xs) {
    int e   = blockIdx.x * blockDim.x + threadIdx.x;   // 0 .. B*S*I-1
    int i   = e & (I_ - 1);
    int tok = e >> 8;                                   // s*32 + b
    int b   = tok & 31;
    int s   = tok >> 5;
    int id  = idx[b * S_ + s];
    xs[e]   = (_Float16)emb[(size_t)id * I_ + i];
}

// ---------------------------------------------------------------------------
// Persistent recurrence kernel: one workgroup, 512 threads = 16 waves.
// h state [32][512] f16 in LDS; per step:
//   pre = x_t @ W_in^T + h @ W_rec^T + b      (WMMA, K=256 then K=512)
//   h_t = tanh(pre)                           -> LDS
//   t2  = h_t @ W_tr^T + b_tr                 (WMMA, K=512)
//   h   = LayerNorm(tanh(t2))                 -> LDS + global hs[B][S][H]
// Wave w owns N-tiles {2w, 2w+1} and both M-tiles (M=32 rows = batch).
// Weights stream from L2 every step (1.25 MB f16 fits L2 trivially); an asm
// memory barrier per step stops the compiler from hoisting+spilling them.
// ---------------------------------------------------------------------------
__launch_bounds__(512, 1)
__global__ void rnn_recur(const _Float16* __restrict__ Win,
                          const _Float16* __restrict__ Wrec,
                          const _Float16* __restrict__ Wtr,
                          const _Float16* __restrict__ xs,
                          const float* __restrict__ bin,
                          const float* __restrict__ brec,
                          const float* __restrict__ btr,
                          const float* __restrict__ gamma,
                          const float* __restrict__ beta,
                          _Float16* __restrict__ hs) {
    __shared__ __align__(16) _Float16 sh_h[B_ * H_];   // 32 KB: current h (f16)
    __shared__ __align__(16) _Float16 sh_t[B_ * H_];   // 32 KB: intermediates

    const int t    = threadIdx.x;
    const int wave = t >> 5;
    const int lane = t & 31;
    const int ln   = lane & 15;          // row/col within a 16-wide tile
    const int half = lane >> 4;          // which half of the wave

    // zero-init h0
    for (int k = t; k < B_ * H_; k += 512) sh_h[k] = (_Float16)0.0f;
    __syncthreads();

    const int n0 = wave * 2;             // this wave's two N tiles
    const int c0 = n0 * 16 + ln;         // hidden column for tile 0
    const int c1 = c0 + 16;              // hidden column for tile 1
    const float bias1_0 = bin[c0] + brec[c0];
    const float bias1_1 = bin[c1] + brec[c1];
    const float bias2_0 = btr[c0];
    const float bias2_1 = btr[c1];

    // LayerNorm pass mapping: 16 threads per batch row
    const int lb = t >> 4;               // batch row 0..31
    const int lj = t & 15;               // 16 chunks of 32 elems

    for (int s = 0; s < S_; ++s) {
        // stop LICM of weight-fragment loads out of the time loop (spill killer)
        asm volatile("" ::: "memory");

        // ---------------- GEMM1: x_t @ W_in^T + h @ W_rec^T ----------------
        v8f acc[2][2] = {};
        const _Float16* xrow = xs + (size_t)s * (B_ * I_);

        // K over I=256 (x part); A rows = batch, straight from global
        for (int kc = 0; kc < I_ / 32; ++kc) {
            const int ko = kc * 32;
            v16h a0 = load_frag(xrow + (size_t)(ln)      * I_ + ko + half * 8,
                                xrow + (size_t)(ln)      * I_ + ko + 16 + half * 8);
            v16h a1 = load_frag(xrow + (size_t)(16 + ln) * I_ + ko + half * 8,
                                xrow + (size_t)(16 + ln) * I_ + ko + 16 + half * 8);
            v16h b0 = load_frag(Win + (size_t)c0 * I_ + ko + half * 8,
                                Win + (size_t)c0 * I_ + ko + 16 + half * 8);
            v16h b1 = load_frag(Win + (size_t)c1 * I_ + ko + half * 8,
                                Win + (size_t)c1 * I_ + ko + 16 + half * 8);
            acc[0][0] = wmma_f16(a0, b0, acc[0][0]);
            acc[0][1] = wmma_f16(a0, b1, acc[0][1]);
            acc[1][0] = wmma_f16(a1, b0, acc[1][0]);
            acc[1][1] = wmma_f16(a1, b1, acc[1][1]);
        }
        // K over H=512 (recurrent part); A from LDS h
        for (int kc = 0; kc < H_ / 32; ++kc) {
            const int ko = kc * 32;
            v16h a0 = load_frag(sh_h + (ln)      * H_ + ko + half * 8,
                                sh_h + (ln)      * H_ + ko + 16 + half * 8);
            v16h a1 = load_frag(sh_h + (16 + ln) * H_ + ko + half * 8,
                                sh_h + (16 + ln) * H_ + ko + 16 + half * 8);
            v16h b0 = load_frag(Wrec + (size_t)c0 * H_ + ko + half * 8,
                                Wrec + (size_t)c0 * H_ + ko + 16 + half * 8);
            v16h b1 = load_frag(Wrec + (size_t)c1 * H_ + ko + half * 8,
                                Wrec + (size_t)c1 * H_ + ko + 16 + half * 8);
            acc[0][0] = wmma_f16(a0, b0, acc[0][0]);
            acc[0][1] = wmma_f16(a0, b1, acc[0][1]);
            acc[1][0] = wmma_f16(a1, b0, acc[1][0]);
            acc[1][1] = wmma_f16(a1, b1, acc[1][1]);
        }

        // tanh -> sh_t (D layout: element r -> M = r + half*8, N = ln)
        for (int mt = 0; mt < 2; ++mt) {
            for (int r = 0; r < 8; ++r) {
                int m = mt * 16 + half * 8 + r;
                sh_t[m * H_ + c0] = (_Float16)tanhf(acc[mt][0][r] + bias1_0);
                sh_t[m * H_ + c1] = (_Float16)tanhf(acc[mt][1][r] + bias1_1);
            }
        }
        __syncthreads();

        // ---------------- GEMM2: h_t @ W_tr^T ----------------
        v8f a2[2][2] = {};
        for (int kc = 0; kc < H_ / 32; ++kc) {
            const int ko = kc * 32;
            v16h a0 = load_frag(sh_t + (ln)      * H_ + ko + half * 8,
                                sh_t + (ln)      * H_ + ko + 16 + half * 8);
            v16h a1 = load_frag(sh_t + (16 + ln) * H_ + ko + half * 8,
                                sh_t + (16 + ln) * H_ + ko + 16 + half * 8);
            v16h b0 = load_frag(Wtr + (size_t)c0 * H_ + ko + half * 8,
                                Wtr + (size_t)c0 * H_ + ko + 16 + half * 8);
            v16h b1 = load_frag(Wtr + (size_t)c1 * H_ + ko + half * 8,
                                Wtr + (size_t)c1 * H_ + ko + 16 + half * 8);
            a2[0][0] = wmma_f16(a0, b0, a2[0][0]);
            a2[0][1] = wmma_f16(a0, b1, a2[0][1]);
            a2[1][0] = wmma_f16(a1, b0, a2[1][0]);
            a2[1][1] = wmma_f16(a1, b1, a2[1][1]);
        }
        __syncthreads();   // all waves done READING sh_t before overwrite

        for (int mt = 0; mt < 2; ++mt) {
            for (int r = 0; r < 8; ++r) {
                int m = mt * 16 + half * 8 + r;
                sh_t[m * H_ + c0] = (_Float16)tanhf(a2[mt][0][r] + bias2_0);
                sh_t[m * H_ + c1] = (_Float16)tanhf(a2[mt][1][r] + bias2_1);
            }
        }
        __syncthreads();

        // ---------------- LayerNorm over H per batch row ----------------
        {
            const _Float16* rowp = sh_t + lb * H_ + lj * 32;
            Frag16 v0, v1;
            v0.q[0] = ((const uint4*)rowp)[0];
            v0.q[1] = ((const uint4*)rowp)[1];
            v1.q[0] = ((const uint4*)rowp)[2];
            v1.q[1] = ((const uint4*)rowp)[3];

            float vals[32];
            float sum = 0.0f, ssq = 0.0f;
            for (int k = 0; k < 16; ++k) { vals[k] = (float)v0.v[k]; vals[16 + k] = (float)v1.v[k]; }
            for (int k = 0; k < 32; ++k) { sum += vals[k]; ssq += vals[k] * vals[k]; }
            // reduce across the 16 lanes that own this row (masks 1..8 stay in group)
            for (int off = 1; off < 16; off <<= 1) {
                sum += __shfl_xor(sum, off, 32);
                ssq += __shfl_xor(ssq, off, 32);
            }
            const float mean = sum * (1.0f / H_);
            const float var  = ssq * (1.0f / H_) - mean * mean;
            const float rinv = rsqrtf(var + LN_EPS);

            Pack32 ou;
            const int hbase = lj * 32;
            for (int k = 0; k < 32; ++k) {
                int   h = hbase + k;
                float y = (vals[k] - mean) * rinv * gamma[h] + beta[h];
                ou.h[k] = (_Float16)y;
            }
            _Float16* dl = sh_h + lb * H_ + hbase;
            ((uint4*)dl)[0] = ou.q[0];
            ((uint4*)dl)[1] = ou.q[1];
            ((uint4*)dl)[2] = ou.q[2];
            ((uint4*)dl)[3] = ou.q[3];
            // hs layout: [B][S][H] so GEMM row m = b*S + s matches output rows
            _Float16* dg = hs + ((size_t)lb * S_ + s) * H_ + hbase;
            ((uint4*)dg)[0] = ou.q[0];
            ((uint4*)dg)[1] = ou.q[1];
            ((uint4*)dg)[2] = ou.q[2];
            ((uint4*)dg)[3] = ou.q[3];
        }
        __syncthreads();
    }
}

// ---------------------------------------------------------------------------
// Output projection: C[M=16384, N=8000] = hs[M,512] @ W_out^T + b_out
// 64x64 tile per block, 4 waves. Tiles staged into LDS by the Tensor Data
// Mover (double-buffered, wave 0 issues, s_wait_tensorcnt + barrier to sync).
// Row m = b*S+s  ==>  out element is just out[m*V + v].
// ---------------------------------------------------------------------------
__launch_bounds__(128, 2)
__global__ void out_gemm(const _Float16* __restrict__ hsm,
                         const _Float16* __restrict__ Wout,
                         const float* __restrict__ bout,
                         float* __restrict__ out) {
    __shared__ __align__(16) _Float16 sA[2][64 * 32];   // 2 x 4 KB
    __shared__ __align__(16) _Float16 sB[2][64 * 32];   // 2 x 4 KB

    const int t     = threadIdx.x;
    const int lane  = t & 31;
    const int wave  = t >> 5;
    const int ln    = lane & 15;
    const int half  = lane >> 4;
    const int mBase = blockIdx.x * 64;
    const int vBase = blockIdx.y * 64;

    v8f acc[4] = {};

    const int NKC = H_ / 32;

    // prologue: TDM load of K-chunk 0 into buffer 0 (wave 0 only; EXEC ignored)
    if (t < 32) {
        tdm_load_2d((unsigned int)(uintptr_t)&sA[0][0],
                    hsm + (size_t)mBase * H_, H_, B_ * S_, 32, 64, H_);
        tdm_load_2d((unsigned int)(uintptr_t)&sB[0][0],
                    Wout + (size_t)vBase * H_, H_, V_, 32, 64, H_);
    }

    for (int kc = 0; kc < NKC; ++kc) {
        const int cur = kc & 1;
        if (t < 32) __builtin_amdgcn_s_wait_tensorcnt(0);
        __syncthreads();                 // buffer `cur` ready for everyone

        // kick off next K-chunk into the other buffer (safe: everyone finished
        // reading it in iteration kc-1 before the barrier above)
        if (t < 32 && kc + 1 < NKC) {
            const int ko = (kc + 1) * 32;
            tdm_load_2d((unsigned int)(uintptr_t)&sA[cur ^ 1][0],
                        hsm + (size_t)mBase * H_ + ko, H_, B_ * S_, 32, 64, H_);
            tdm_load_2d((unsigned int)(uintptr_t)&sB[cur ^ 1][0],
                        Wout + (size_t)vBase * H_ + ko, H_, V_, 32, 64, H_);
        }

        v16h a = load_frag(&sA[cur][(wave * 16 + ln) * 32 + half * 8],
                           &sA[cur][(wave * 16 + ln) * 32 + 16 + half * 8]);
        for (int nt = 0; nt < 4; ++nt) {
            v16h b = load_frag(&sB[cur][(nt * 16 + ln) * 32 + half * 8],
                               &sB[cur][(nt * 16 + ln) * 32 + 16 + half * 8]);
            acc[nt] = wmma_f16(a, b, acc[nt]);
        }
        __syncthreads();                 // done reading `cur` before it is refilled
    }

    // epilogue: acc element r -> M = r + half*8 (within strip), N = ln (within tile)
    for (int nt = 0; nt < 4; ++nt) {
        const int v   = vBase + nt * 16 + ln;
        const float bo = bout[v];
        for (int r = 0; r < 8; ++r) {
            int m = mBase + wave * 16 + half * 8 + r;   // m = b*S + s
            out[(size_t)m * V_ + v] = acc[nt][r] + bo;
        }
    }
}

// ---------------------------------------------------------------------------
// Host-side launch
// ---------------------------------------------------------------------------
extern "C" void kernel_launch(void* const* d_in, const int* in_sizes, int n_in,
                              void* d_out, int out_size, void* d_ws, size_t ws_size,
                              hipStream_t stream) {
    (void)in_sizes; (void)n_in; (void)out_size; (void)ws_size;

    const int*   x_idx = (const int*)  d_in[0];
    const float* emb   = (const float*)d_in[1];
    const float* W_in  = (const float*)d_in[2];
    const float* b_in  = (const float*)d_in[3];
    const float* W_rec = (const float*)d_in[4];
    const float* b_rec = (const float*)d_in[5];
    const float* W_tr  = (const float*)d_in[6];
    const float* b_tr  = (const float*)d_in[7];
    const float* gamma = (const float*)d_in[8];
    const float* beta  = (const float*)d_in[9];
    const float* W_out = (const float*)d_in[10];
    const float* b_out = (const float*)d_in[11];
    float* out = (float*)d_out;

    // workspace layout (bytes)
    char* ws = (char*)d_ws;
    _Float16* Win_h  = (_Float16*)(ws + 0);           //  512*256*2 = 262144
    _Float16* Wrec_h = (_Float16*)(ws + 262144);      //  512*512*2 = 524288
    _Float16* Wtr_h  = (_Float16*)(ws + 786432);      //  512*512*2 = 524288
    _Float16* Wout_h = (_Float16*)(ws + 1310720);     // 8000*512*2 = 8192000
    _Float16* xs_h   = (_Float16*)(ws + 9502720);     //  512*32*256*2 = 8388608
    _Float16* hs_h   = (_Float16*)(ws + 17891328);    //  512*32*512*2 = 16777216
    // total: 34,668,544 bytes

    cvt_f32_f16<<<(H_ * I_ + 255) / 256, 256, 0, stream>>>(W_in,  Win_h,  H_ * I_);
    cvt_f32_f16<<<(H_ * H_ + 255) / 256, 256, 0, stream>>>(W_rec, Wrec_h, H_ * H_);
    cvt_f32_f16<<<(H_ * H_ + 255) / 256, 256, 0, stream>>>(W_tr,  Wtr_h,  H_ * H_);
    cvt_f32_f16<<<(V_ * H_ + 255) / 256, 256, 0, stream>>>(W_out, Wout_h, V_ * H_);

    gather_emb<<<(B_ * S_ * I_) / 256, 256, 0, stream>>>(x_idx, emb, xs_h);

    rnn_recur<<<1, 512, 0, stream>>>(Win_h, Wrec_h, Wtr_h, xs_h,
                                     b_in, b_rec, b_tr, gamma, beta, hs_h);

    out_gemm<<<dim3((B_ * S_) / 64, V_ / 64), 128, 0, stream>>>(hs_h, Wout_h, b_out, out);
}
